// DiffeqSolver_64587718197724
// MI455X (gfx1250) — matrix operand order repaired and checked by reading, not compiled, not run on hardware
//
#include <hip/hip_runtime.h>
#include <hip/hip_bf16.h>
#include <math.h>

typedef _Float16 h16;
typedef __attribute__((ext_vector_type(16))) _Float16 v16h;
typedef __attribute__((ext_vector_type(8)))  _Float16 v8h;
typedef __attribute__((ext_vector_type(8)))  float    v8f;
typedef __attribute__((ext_vector_type(4)))  int      i32x4;
typedef __attribute__((address_space(1)))    i32x4    gi32x4;  // global
typedef __attribute__((address_space(3)))    i32x4    li32x4;  // LDS

#define N_TRAJ 4
#define N_B    256
#define N_D    256
#define N_H    1024
#define N_T    50
#define M_ROWS 1024   // N_TRAJ * N_B

// CDNA5 async global->LDS staging path (ASYNCcnt-tracked), with safe fallback.
#if __has_builtin(__builtin_amdgcn_global_load_async_to_lds_b128) && \
    __has_builtin(__builtin_amdgcn_s_wait_asynccnt)
#define USE_ASYNC_LDS 1
#else
#define USE_ASYNC_LDS 0
#endif

// ---------------------------------------------------------------------------
// Weight convert + transpose: W[K][N] f32 (row-major) -> Wt[N][K] f16.
// ---------------------------------------------------------------------------
__global__ void k_transpose_f16(const float* __restrict__ W, h16* __restrict__ Wt,
                                int K, int N) {
  int idx = blockIdx.x * blockDim.x + threadIdx.x;
  if (idx >= K * N) return;
  int k = idx / N, n = idx - k * N;
  Wt[(size_t)n * K + k] = (h16)W[idx];
}

// ---------------------------------------------------------------------------
// Init: y32/yin16 <- first_point, and write out[:, :, t=0, :].
// ---------------------------------------------------------------------------
__global__ void k_init(const float* __restrict__ fp, float* __restrict__ y32,
                       h16* __restrict__ yin16, float* __restrict__ out) {
  int i = blockIdx.x * blockDim.x + threadIdx.x;   // [0, M_ROWS*N_D)
  float v = fp[i];
  y32[i]   = v;
  yin16[i] = (h16)v;
  int m = i / N_D, d = i - m * N_D;
  out[((size_t)m * N_T + 0) * N_D + d] = v;
}

// ---------------------------------------------------------------------------
// WMMA fragment layouts (wave32, V_WMMA_F32_16X16X32_F16, ISA 7.12.2)
// A 16x32 f16: lane<16: row M=lane,   K = kk+0..7  & kk+16..23
//              lane>=16: row M=lane-16, K = kk+8..15 & kk+24..31
// B 32x16 f16 (col n row-contiguous): lane<16: col=lane, K=kk+0..7 & kk+16..23
//                                     lane>=16: col=lane-16, K=kk+8..15 & kk+24..31
// ---------------------------------------------------------------------------
__device__ __forceinline__ v16h load_a_frag(const h16* __restrict__ rowp, int kk, int lhalf) {
  const h16* p = rowp + kk + (lhalf << 3);
  v8h lo = *(const v8h*)(p);
  v8h hi = *(const v8h*)(p + 16);
  return __builtin_shufflevector(lo, hi, 0,1,2,3,4,5,6,7,8,9,10,11,12,13,14,15);
}

// Stage one 32-K chunk of B (BN weight rows x 32 halves) into LDS.
// LDS layout: row r of chunk at dst[r*32 .. r*32+31] (halves).
template<int KTOT, int BN>
__device__ __forceinline__ void stage_b(const h16* __restrict__ Wt, h16* dst,
                                        int col0, int kk, int tid) {
  constexpr int SLOTS = BN * 4;          // 16-byte slots in one chunk buffer
#pragma unroll
  for (int i = 0; i < SLOTS / 256; ++i) {
    int s = tid + i * 256;
    int r = s >> 2;                      // weight row within [0, BN)
    int c = (s & 3) << 3;                // half offset within the 32-K chunk
    const h16* g = Wt + (size_t)(col0 + r) * KTOT + kk + c;
    h16* l = dst + r * 32 + c;
#if USE_ASYNC_LDS
    __builtin_amdgcn_global_load_async_to_lds_b128(
        (gi32x4*)g, (li32x4*)l, 0, 0);
#else
    *(v8h*)l = *(const v8h*)g;
#endif
  }
}

// Core: block tile = 128(M) x BN(N); wave owns 16 rows x BN cols (NT=BN/16
// wmma tiles). B double-buffered through LDS; A streamed from global.
// NOTE: chunk loop kept at unroll 1 so the accumulator tuples stay pinned in
// one register set (unroll-2 made the allocator rotate all acc VGPRs through
// v_mov_b64 each iteration, adding ~1 VALU op + hazard nops per WMMA).
template<int KTOT, int BN>
__device__ __forceinline__ void gemm_core(const h16* __restrict__ A,
                                          const h16* __restrict__ Wt,
                                          h16* smem, int row0, int col0,
                                          int tid, v8f* acc) {
  constexpr int NT  = BN / 16;
  constexpr int BUF = BN * 32;           // halves per chunk buffer
  constexpr int NC  = KTOT / 32;
  const int lane = tid & 31;
  const int lmod = lane & 15, lhalf = lane >> 4;
  const h16* arow = A + (size_t)(row0 + lmod) * KTOT;

  stage_b<KTOT, BN>(Wt, smem, col0, 0, tid);
#pragma unroll 1
  for (int c = 0; c < NC; ++c) {
#if USE_ASYNC_LDS
    __builtin_amdgcn_s_wait_asynccnt(0);
#endif
    __syncthreads();                      // staged chunk visible; prev reads done
    if (c + 1 < NC)
      stage_b<KTOT, BN>(Wt, smem + ((c + 1) & 1) * BUF, col0, (c + 1) * 32, tid);
    const h16* bbuf = smem + (c & 1) * BUF;
    v16h a = load_a_frag(arow, c * 32, lhalf);
#pragma unroll
    for (int t = 0; t < NT; ++t) {
      const h16* p = bbuf + (t * 16 + lmod) * 32 + (lhalf << 3);
      v8h lo = *(const v8h*)p;
      v8h hi = *(const v8h*)(p + 16);
      v16h b = __builtin_shufflevector(lo, hi, 0,1,2,3,4,5,6,7,8,9,10,11,12,13,14,15);
      acc[t] = __builtin_amdgcn_wmma_f32_16x16x32_f16(
          false, a, false, b, (short)0, acc[t], false, false);
    }
  }
}

// ---------------------------------------------------------------------------
// GEMM + bias + tanh, f16 output (layers 1 and 2). Block 256 thr = 8 waves,
// block tile 128 x 128.
// ---------------------------------------------------------------------------
template<int KTOT>
__global__ void k_gemm_tanh(const h16* __restrict__ A, const h16* __restrict__ Wt,
                            const float* __restrict__ bias, h16* __restrict__ Out,
                            int N) {
  constexpr int BN = 128;
  __shared__ h16 smem[2 * BN * 32];      // 16 KB
  const int tid  = threadIdx.x;
  const int wave = tid >> 5, lane = tid & 31;
  const int lmod = lane & 15, lhalf = lane >> 4;
  const int row0 = blockIdx.x * 128 + wave * 16;
  const int col0 = blockIdx.y * BN;
  v8f acc[BN / 16] = {};
  gemm_core<KTOT, BN>(A, Wt, smem, row0, col0, tid, acc);
#pragma unroll
  for (int t = 0; t < BN / 16; ++t) {
    int col = col0 + t * 16 + lmod;
    float bv = bias[col];
#pragma unroll
    for (int v = 0; v < 8; ++v) {
      int row = row0 + v + (lhalf << 3);
      Out[(size_t)row * N + col] = (h16)tanhf(acc[t][v] + bv);
    }
  }
}

// ---------------------------------------------------------------------------
// GEMM3 (+bias) fused with the RK4 stage update. N = N_D, K = N_H, BN = 64.
//   stage 0: acc32  = (dt/6)*k1;  yin16 = f16(y + 0.5*dt*k1)
//   stage 1: acc32 += (dt/3)*k2;  yin16 = f16(y + 0.5*dt*k2)
//   stage 2: acc32 += (dt/3)*k3;  yin16 = f16(y +     dt*k3)
//   stage 3: y' = y + acc32 + (dt/6)*k4; write y32, yin16, out[:,:,step+1,:]
// ---------------------------------------------------------------------------
__global__ void k_gemm_rk4(const h16* __restrict__ A, const h16* __restrict__ Wt,
                           const float* __restrict__ bias, const float* __restrict__ ts,
                           int step, int stage,
                           float* __restrict__ y32, float* __restrict__ acc32,
                           h16* __restrict__ yin16, float* __restrict__ out) {
  constexpr int BN = 64;
  __shared__ h16 smem[2 * BN * 32];      // 8 KB
  const int tid  = threadIdx.x;
  const int wave = tid >> 5, lane = tid & 31;
  const int lmod = lane & 15, lhalf = lane >> 4;
  const int row0 = blockIdx.x * 128 + wave * 16;
  const int col0 = blockIdx.y * BN;
  v8f acc[BN / 16] = {};
  gemm_core<N_H, BN>(A, Wt, smem, row0, col0, tid, acc);

  const float dt = ts[step + 1] - ts[step];      // uniform scalar loads
  const float w6 = dt * (1.0f / 6.0f);
  const float w3 = dt * (1.0f / 3.0f);
#pragma unroll
  for (int t = 0; t < BN / 16; ++t) {
    int col = col0 + t * 16 + lmod;
    float bv = bias[col];
#pragma unroll
    for (int v = 0; v < 8; ++v) {
      int row = row0 + v + (lhalf << 3);
      size_t idx = (size_t)row * N_D + col;
      float kv = acc[t][v] + bv;
      if (stage == 0) {
        acc32[idx] = w6 * kv;
        yin16[idx] = (h16)(y32[idx] + 0.5f * dt * kv);
      } else if (stage == 1) {
        acc32[idx] += w3 * kv;
        yin16[idx] = (h16)(y32[idx] + 0.5f * dt * kv);
      } else if (stage == 2) {
        acc32[idx] += w3 * kv;
        yin16[idx] = (h16)(y32[idx] + dt * kv);
      } else {
        float yn = y32[idx] + acc32[idx] + w6 * kv;
        y32[idx]   = yn;
        yin16[idx] = (h16)yn;
        out[((size_t)row * N_T + (step + 1)) * N_D + col] = yn;
      }
    }
  }
}

// ---------------------------------------------------------------------------
extern "C" void kernel_launch(void* const* d_in, const int* in_sizes, int n_in,
                              void* d_out, int out_size, void* d_ws, size_t ws_size,
                              hipStream_t stream) {
  (void)in_sizes; (void)n_in; (void)out_size; (void)ws_size;
  const float* fp = (const float*)d_in[0];   // first_point   [4,256,256]
  const float* ts = (const float*)d_in[1];   // times         [50]
  const float* W1 = (const float*)d_in[2];   // [256,1024]
  const float* b1 = (const float*)d_in[3];
  const float* W2 = (const float*)d_in[4];   // [1024,1024]
  const float* b2 = (const float*)d_in[5];
  const float* W3 = (const float*)d_in[6];   // [1024,256]
  const float* b3 = (const float*)d_in[7];
  float* out = (float*)d_out;                // [4,256,50,256]

  // Workspace carve (~10 MB total)
  char* p = (char*)d_ws;
  auto carve = [&](size_t bytes) -> char* {
    char* r = p; p += (bytes + 255) & ~(size_t)255; return r;
  };
  float* y32   = (float*)carve((size_t)M_ROWS * N_D * 4);
  float* acc32 = (float*)carve((size_t)M_ROWS * N_D * 4);
  h16*   yin16 = (h16*)  carve((size_t)M_ROWS * N_D * 2);
  h16*   h1    = (h16*)  carve((size_t)M_ROWS * N_H * 2);
  h16*   h2    = (h16*)  carve((size_t)M_ROWS * N_H * 2);
  h16*   Wt1   = (h16*)  carve((size_t)N_H * N_D * 2);   // [H, D]
  h16*   Wt2   = (h16*)  carve((size_t)N_H * N_H * 2);   // [H, H]
  h16*   Wt3   = (h16*)  carve((size_t)N_D * N_H * 2);   // [D, H]

  k_transpose_f16<<<(N_D * N_H + 255) / 256, 256, 0, stream>>>(W1, Wt1, N_D, N_H);
  k_transpose_f16<<<(N_H * N_H + 255) / 256, 256, 0, stream>>>(W2, Wt2, N_H, N_H);
  k_transpose_f16<<<(N_H * N_D + 255) / 256, 256, 0, stream>>>(W3, Wt3, N_H, N_D);
  k_init<<<(M_ROWS * N_D) / 256, 256, 0, stream>>>(fp, y32, yin16, out);

  dim3 blk(256);
  dim3 gH(M_ROWS / 128, N_H / 128);   // (8,8) for H-wide layers
  dim3 gD(M_ROWS / 128, N_D / 64);    // (8,4) for the RK4 layer
  for (int step = 0; step < N_T - 1; ++step) {
    for (int stage = 0; stage < 4; ++stage) {
      k_gemm_tanh<N_D><<<gH, blk, 0, stream>>>(yin16, Wt1, b1, h1, N_H);
      k_gemm_tanh<N_H><<<gH, blk, 0, stream>>>(h1,    Wt2, b2, h2, N_H);
      k_gemm_rk4       <<<gD, blk, 0, stream>>>(h2, Wt3, b3, ts, step, stage,
                                                y32, acc32, yin16, out);
    }
  }
}
